// SelfAttn_52613349376684
// MI455X (gfx1250) — matrix operand order repaired and checked by reading, not compiled, not run on hardware
//
#include <hip/hip_runtime.h>

#define DEVINL __device__ __forceinline__

typedef __attribute__((ext_vector_type(16))) __bf16 v16bf;
typedef __attribute__((ext_vector_type(8)))  __bf16 v8bf;
typedef __attribute__((ext_vector_type(8)))  float  v8f;

// D = A(16x32 bf16) * B(32x16 bf16) + C(16x16 f32)
#define WMMA_BF16(acc, a, b)                                                   \
  (acc) = __builtin_amdgcn_wmma_f32_16x16x32_bf16(false, (a), false, (b),      \
                                                  (short)0, (acc), false, false)

// A-operand (16x32, bf16): lane<16 holds row M=lane, K elems {0..7,16..23};
// lane>=16 holds row M=lane-16, K elems {8..15,24..31}. rowk = &A[row][kbase].
DEVINL v16bf ld_a32(const __bf16* rowk, int lane) {
  const int off = (lane >> 4) << 3;            // 0 or 8
  v8bf lo = *(const v8bf*)(rowk + off);        // K = off .. off+7
  v8bf hi = *(const v8bf*)(rowk + off + 16);   // K = off+16 .. off+23
  return __builtin_shufflevector(lo, hi, 0, 1, 2, 3, 4, 5, 6, 7, 8, 9, 10, 11,
                                 12, 13, 14, 15);
}
// B-operand (32x16, bf16): lane holds column N=lane&15, K elems {0..7,16..23}
// (lane<16) or {8..15,24..31} (lane>=16), matching the A operand.
// colk = &Bt[col][kbase] where Bt is K-contiguous per column.
DEVINL v16bf ld_b32(const __bf16* colk, int lane) {
  const int off = (lane >> 4) << 3;
  v8bf lo = *(const v8bf*)(colk + off);
  v8bf hi = *(const v8bf*)(colk + off + 16);
  return __builtin_shufflevector(lo, hi, 0, 1, 2, 3, 4, 5, 6, 7, 8, 9, 10, 11,
                                 12, 13, 14, 15);
}

// ---------------- conversion / layout kernels ----------------
__global__ void k_cvt(const float* __restrict__ in, __bf16* __restrict__ out,
                      int n) {
  int i = blockIdx.x * blockDim.x + threadIdx.x;
  if (i < n) out[i] = (__bf16)in[i];
}

// in[r][c] (rows x cols) f32 -> out[c][r] bf16
__global__ void k_tcvt(const float* __restrict__ in, __bf16* __restrict__ out,
                       int rows, int cols) {
  int i = blockIdx.x * blockDim.x + threadIdx.x;
  if (i >= rows * cols) return;
  int r = i / cols, c = i % cols;
  out[(size_t)c * rows + r] = (__bf16)in[i];
}

// pe (1024,1024,32) f32 -> pe_t (1024,32,1024) bf16
__global__ void k_pet(const float* __restrict__ pe, __bf16* __restrict__ pet) {
  int i = blockIdx.x * blockDim.x + threadIdx.x; // 1024*32*1024 threads
  int j = i & 1023;
  int c = (i >> 10) & 31;
  int ir = i >> 15;
  pet[i] = (__bf16)pe[((size_t)ir * 1024 + j) * 32 + c];
}

// ---------------- phase 2: QKV projection ----------------
// x_bf (4096x256) @ W (256x512); Q,K stored (B,H,N,64); V stored transposed
// (B,H,64,N) with bias added. One wave per 16x16 output tile, 3 matrices.
__global__ void k_qkv(const __bf16* __restrict__ xb,
                      const __bf16* __restrict__ wqT,
                      const __bf16* __restrict__ wkT,
                      const __bf16* __restrict__ wvT,
                      const float* __restrict__ bv, __bf16* __restrict__ Q,
                      __bf16* __restrict__ K, __bf16* __restrict__ Vt) {
  const int lane = threadIdx.x & 31;
  const int wid = blockIdx.x * (blockDim.x >> 5) + (threadIdx.x >> 5);
  const int which = wid / (256 * 32);
  const int rem = wid % (256 * 32);
  const int mt = rem >> 5;  // 256 row tiles over 4096 x-rows
  const int et = rem & 31;  // 32 col tiles over E=512
  const __bf16* wT = (which == 0) ? wqT : (which == 1) ? wkT : wvT;
  const int half = lane >> 4, lidx = lane & 15;
  const __bf16* arow = xb + (size_t)(mt * 16 + lidx) * 256;
  const __bf16* bcol = wT + (size_t)(et * 16 + lidx) * 256;
  v8f acc = {};
#pragma unroll
  for (int kb = 0; kb < 256; kb += 32) {
    v16bf a = ld_a32(arow + kb, lane);
    v16bf b = ld_b32(bcol + kb, lane);
    WMMA_BF16(acc, a, b);
  }
  const int e = et * 16 + lidx;  // lane's output channel
  const int h = e >> 6, d = e & 63;
  if (which < 2) {
    __bf16* dst = (which == 0) ? Q : K;
#pragma unroll
    for (int r = 0; r < 8; r++) {
      const int m = mt * 16 + half * 8 + r;  // = b*1024 + n
      const int bb = m >> 10, n = m & 1023;
      dst[(((size_t)(bb * 8 + h)) * 1024 + n) * 64 + d] = (__bf16)acc[r];
    }
  } else {
    const float bias = bv[e];
#pragma unroll
    for (int r = 0; r < 8; r++) {
      const int m = mt * 16 + half * 8 + r;
      const int bb = m >> 10, n = m & 1023;
      Vt[(((size_t)(bb * 8 + h)) * 64 + d) * 1024 + n] = (__bf16)(acc[r] + bias);
    }
  }
}

// ---------------- phase 3: scores + exp + row sums ----------------
// One wave per (bh, 16-row i tile): S = scale*Q K^T, P = exp(S) (unnormalized,
// scores are O(1) so no max subtraction needed), rowsum accumulated.
__global__ void k_score(const __bf16* __restrict__ Q,
                        const __bf16* __restrict__ K, __bf16* __restrict__ P,
                        float* __restrict__ rowsum) {
  const int lane = threadIdx.x & 31;
  const int wid = blockIdx.x * (blockDim.x >> 5) + (threadIdx.x >> 5);
  const int bh = wid >> 6;  // 32
  const int it = wid & 63;  // 64 i-tiles
  const int half = lane >> 4, lidx = lane & 15;
  const __bf16* arow = Q + ((size_t)bh * 1024 + it * 16 + lidx) * 64;
  float rs[8];
#pragma unroll
  for (int r = 0; r < 8; r++) rs[r] = 0.f;
  for (int jt = 0; jt < 64; jt++) {
    const __bf16* bcol = K + ((size_t)bh * 1024 + jt * 16 + lidx) * 64;
    v8f acc = {};
#pragma unroll
    for (int kb = 0; kb < 64; kb += 32) {
      v16bf a = ld_a32(arow + kb, lane);
      v16bf b = ld_b32(bcol + kb, lane);
      WMMA_BF16(acc, a, b);
    }
#pragma unroll
    for (int r = 0; r < 8; r++) {
      const float ex = __expf(acc[r] * 0.125f);  // scale = 64^-0.5
      rs[r] += ex;
      const int i = it * 16 + half * 8 + r;
      P[(size_t)bh * (1024u * 1024u) + (size_t)i * 1024 + jt * 16 + lidx] =
          (__bf16)ex;
    }
  }
  // reduce row sums across the 16 lanes of each half-wave
#pragma unroll
  for (int m = 1; m < 16; m <<= 1) {
#pragma unroll
    for (int r = 0; r < 8; r++) rs[r] += __shfl_xor(rs[r], m, 32);
  }
  if (lidx == 0) {
#pragma unroll
    for (int r = 0; r < 8; r++)
      rowsum[bh * 1024 + it * 16 + half * 8 + r] = rs[r];
  }
}

// ---------------- phase 4: O = (P @ V) / rowsum ----------------
__global__ void k_pv(const __bf16* __restrict__ P, const __bf16* __restrict__ Vt,
                     const float* __restrict__ rowsum, float* __restrict__ O) {
  const int lane = threadIdx.x & 31;
  const int wid = blockIdx.x * (blockDim.x >> 5) + (threadIdx.x >> 5);
  const int bh = wid >> 6, it = wid & 63;
  const int half = lane >> 4, lidx = lane & 15;
  const __bf16* arow =
      P + (size_t)bh * (1024u * 1024u) + (size_t)(it * 16 + lidx) * 1024;
  v8f acc[4] = {};
  for (int kb = 0; kb < 1024; kb += 32) {
    __builtin_prefetch(arow + kb + 128, 0, 0);
    v16bf a = ld_a32(arow + kb, lane);
#pragma unroll
    for (int dt = 0; dt < 4; dt++) {
      v16bf b = ld_b32(Vt + ((size_t)bh * 64 + dt * 16 + lidx) * 1024 + kb, lane);
      WMMA_BF16(acc[dt], a, b);
    }
  }
  float rinv[8];
#pragma unroll
  for (int r = 0; r < 8; r++)
    rinv[r] = 1.0f / rowsum[bh * 1024 + it * 16 + half * 8 + r];
  const int bb = bh >> 3, h = bh & 7;
#pragma unroll
  for (int dt = 0; dt < 4; dt++) {
#pragma unroll
    for (int r = 0; r < 8; r++) {
      const int i = it * 16 + half * 8 + r;
      const int d = dt * 16 + lidx;
      O[((size_t)(bb * 1024 + i)) * 512 + h * 64 + d] = acc[dt][r] * rinv[r];
    }
  }
}

// ---------------- phase 5: pe_sum (per fixed i: 32bh x 1024j x 32c GEMM) ----
__global__ void k_pesum(const __bf16* __restrict__ P,
                        const __bf16* __restrict__ peT,
                        const float* __restrict__ rowsum,
                        __bf16* __restrict__ PS) {
  const int lane = threadIdx.x & 31;
  const int i = blockIdx.x * (blockDim.x >> 5) + (threadIdx.x >> 5);  // 1024
  const int half = lane >> 4, lidx = lane & 15;
  const __bf16* a0row = P + (size_t)lidx * (1024u * 1024u) + (size_t)i * 1024;
  const __bf16* a1row =
      P + (size_t)(16 + lidx) * (1024u * 1024u) + (size_t)i * 1024;
  v8f acc00 = {}, acc01 = {}, acc10 = {}, acc11 = {};
  for (int kb = 0; kb < 1024; kb += 32) {
    v16bf a0 = ld_a32(a0row + kb, lane);
    v16bf a1 = ld_a32(a1row + kb, lane);
    v16bf b0 = ld_b32(peT + ((size_t)i * 32 + lidx) * 1024 + kb, lane);
    v16bf b1 = ld_b32(peT + ((size_t)i * 32 + 16 + lidx) * 1024 + kb, lane);
    WMMA_BF16(acc00, a0, b0);
    WMMA_BF16(acc01, a0, b1);
    WMMA_BF16(acc10, a1, b0);
    WMMA_BF16(acc11, a1, b1);
  }
#pragma unroll
  for (int r = 0; r < 8; r++) {
    const int bh0 = half * 8 + r;
    const int bh1 = 16 + bh0;
    const float inv0 = 1.0f / rowsum[bh0 * 1024 + i];
    const float inv1 = 1.0f / rowsum[bh1 * 1024 + i];
    PS[((size_t)bh0 * 1024 + i) * 32 + lidx] = (__bf16)(acc00[r] * inv0);
    PS[((size_t)bh0 * 1024 + i) * 32 + 16 + lidx] = (__bf16)(acc01[r] * inv0);
    PS[((size_t)bh1 * 1024 + i) * 32 + lidx] = (__bf16)(acc10[r] * inv1);
    PS[((size_t)bh1 * 1024 + i) * 32 + 16 + lidx] = (__bf16)(acc11[r] * inv1);
  }
}

// ---------------- phase 6: O += PS @ Wpe + bpe; emit bf16 ----------------
__global__ void k_peproj(const __bf16* __restrict__ PS,
                         const __bf16* __restrict__ wpeT,
                         const float* __restrict__ bpe,
                         const float* __restrict__ O,
                         __bf16* __restrict__ Obf) {
  const int lane = threadIdx.x & 31;
  const int wid = blockIdx.x * (blockDim.x >> 5) + (threadIdx.x >> 5);  // 8192
  const int mt = wid >> 2, dt = wid & 3;
  const int half = lane >> 4, lidx = lane & 15;
  v16bf a = ld_a32(PS + (size_t)(mt * 16 + lidx) * 32, lane);  // K=32 exactly
  v16bf b = ld_b32(wpeT + (size_t)(dt * 16 + lidx) * 32, lane);
  v8f acc = {};
  WMMA_BF16(acc, a, b);
  const int d = dt * 16 + lidx;
  const float bias = bpe[d];
#pragma unroll
  for (int r = 0; r < 8; r++) {
    const int m = mt * 16 + half * 8 + r;  // m = bh*1024 + i
    const int bh = m >> 10, i = m & 1023;
    const int bb = bh >> 3, h = bh & 7;
    const size_t o = ((size_t)(bb * 1024 + i)) * 512 + h * 64 + d;
    Obf[o] = (__bf16)(O[o] + acc[r] + bias);
  }
}

// ---------------- phase 7: out = Obf @ Wproj ----------------
__global__ void k_proj(const __bf16* __restrict__ Obf,
                       const __bf16* __restrict__ wpT,
                       float* __restrict__ out) {
  const int lane = threadIdx.x & 31;
  const int wid = blockIdx.x * (blockDim.x >> 5) + (threadIdx.x >> 5);  // 4096
  const int mt = wid >> 4, nt = wid & 15;
  const int half = lane >> 4, lidx = lane & 15;
  const __bf16* arow = Obf + (size_t)(mt * 16 + lidx) * 512;
  const __bf16* bcol = wpT + (size_t)(nt * 16 + lidx) * 512;
  v8f acc = {};
#pragma unroll
  for (int kb = 0; kb < 512; kb += 32) {
    __builtin_prefetch(arow + kb + 64, 0, 0);
    v16bf a = ld_a32(arow + kb, lane);
    v16bf b = ld_b32(bcol + kb, lane);
    WMMA_BF16(acc, a, b);
  }
#pragma unroll
  for (int r = 0; r < 8; r++) {
    const int m = mt * 16 + half * 8 + r;
    out[(size_t)m * 256 + nt * 16 + lidx] = acc[r];
  }
}

extern "C" void kernel_launch(void* const* d_in, const int* in_sizes, int n_in,
                              void* d_out, int out_size, void* d_ws,
                              size_t ws_size, hipStream_t stream) {
  const float* x = (const float*)d_in[0];
  const float* pe = (const float*)d_in[1];
  const float* Wq = (const float*)d_in[2];
  const float* Wk = (const float*)d_in[3];
  const float* Wv = (const float*)d_in[4];
  const float* bv = (const float*)d_in[5];
  const float* Wpe = (const float*)d_in[6];
  const float* bpe = (const float*)d_in[7];
  const float* Wproj = (const float*)d_in[8];
  float* out = (float*)d_out;
  (void)in_sizes; (void)n_in; (void)out_size; (void)ws_size;

  char* w = (char*)d_ws;
  size_t off = 0;
  auto take = [&](size_t bytes) -> char* {
    char* p = w + off;
    off += (bytes + 255) & ~((size_t)255);
    return p;
  };
  __bf16* xb   = (__bf16*)take((size_t)4096 * 256 * 2);        // x in bf16
  __bf16* wqT  = (__bf16*)take((size_t)512 * 256 * 2);         // Wq^T
  __bf16* wkT  = (__bf16*)take((size_t)512 * 256 * 2);         // Wk^T
  __bf16* wvT  = (__bf16*)take((size_t)512 * 256 * 2);         // Wv^T
  __bf16* wpeT = (__bf16*)take((size_t)64 * 32 * 2);           // Wpe^T
  __bf16* wpT  = (__bf16*)take((size_t)256 * 512 * 2);         // Wproj^T
  __bf16* peT  = (__bf16*)take((size_t)1024 * 32 * 1024 * 2);  // pe (i,c,j)
  __bf16* Qb   = (__bf16*)take((size_t)32 * 1024 * 64 * 2);    // (bh,n,d)
  __bf16* Kb   = (__bf16*)take((size_t)32 * 1024 * 64 * 2);    // (bh,n,d)
  __bf16* Vt   = (__bf16*)take((size_t)32 * 64 * 1024 * 2);    // (bh,d,n)
  __bf16* P    = (__bf16*)take((size_t)32 * 1024 * 1024 * 2);  // exp scores
  float* rsum  = (float*)take((size_t)32 * 1024 * 4);          // row sums
  float* O     = (float*)take((size_t)4096 * 512 * 4);         // P@V / sum
  __bf16* PS   = (__bf16*)take((size_t)32768 * 32 * 2);        // pe_sum bf16
  __bf16* Obf  = (__bf16*)take((size_t)4096 * 512 * 2);        // O + pe part

  dim3 blk(256);
  k_cvt<<<(4096 * 256 + 255) / 256, blk, 0, stream>>>(x, xb, 4096 * 256);
  k_tcvt<<<(256 * 512 + 255) / 256, blk, 0, stream>>>(Wq, wqT, 256, 512);
  k_tcvt<<<(256 * 512 + 255) / 256, blk, 0, stream>>>(Wk, wkT, 256, 512);
  k_tcvt<<<(256 * 512 + 255) / 256, blk, 0, stream>>>(Wv, wvT, 256, 512);
  k_tcvt<<<(32 * 64 + 255) / 256, blk, 0, stream>>>(Wpe, wpeT, 32, 64);
  k_tcvt<<<(512 * 256 + 255) / 256, blk, 0, stream>>>(Wproj, wpT, 512, 256);
  k_pet<<<(1024 * 32 * 1024) / 256, blk, 0, stream>>>(pe, peT);

  k_qkv<<<3072, blk, 0, stream>>>(xb, wqT, wkT, wvT, bv, Qb, Kb, Vt); // 24576 waves
  k_score<<<256, blk, 0, stream>>>(Qb, Kb, P, rsum);                  // 2048 waves
  k_pv<<<256, blk, 0, stream>>>(P, Vt, rsum, O);                      // 2048 waves
  k_pesum<<<128, blk, 0, stream>>>(P, peT, rsum, PS);                 // 1024 waves
  k_peproj<<<1024, blk, 0, stream>>>(PS, wpeT, bpe, O, Obf);          // 8192 waves
  k_proj<<<512, blk, 0, stream>>>(Obf, wpT, out);                     // 4096 waves
}
